// LSTMModel_15788299780142
// MI455X (gfx1250) — compile-verified, matrix operand order checked by reading, not run on hardware
//
#include <hip/hip_runtime.h>

// ---------------------------------------------------------------------------
// Fused 3-layer LSTM (H=50) + dense head for MI455X (gfx1250, wave32, WMMA).
//
// One workgroup = one 16-row batch tile, persistent across all T=1024 steps
// and all 3 layers (stacked-LSTM chaining is per-timestep, so all three
// layers advance together each step). All recurrent state (h, c) lives in
// LDS, all weights live in VGPRs as pre-packed WMMA B-fragments, and the
// only global traffic inside the scan is a register-pipelined scalar x[b][t].
//
// Gate GEMM per layer per step:  [16 x 128] x [128 x 256] f16 WMMA:
//   A = [ h_pad(64) | x_pad(64) ]          (LDS, f16)
//   B = [ W_hh^T ; W_ih^T ] zero-padded    (registers, f16 fragments)
//   C/D = gate pre-activations             (f32 accum) ; N = 4 gates x 64
// Layer 1 uses only 3 K-steps (its input dim is 1 -> K-step 3 is all-zero).
// ---------------------------------------------------------------------------

typedef __attribute__((ext_vector_type(16))) _Float16 v16h;
typedef __attribute__((ext_vector_type(8)))  _Float16 v8h;
typedef __attribute__((ext_vector_type(8)))  float    v8f;

#define HID    50
#define T_LEN  1024
#define BTILE  16
#define ASTR   136   // A-stage row stride in halfs (128 + 8 bank skew, 16B-aligned rows)
#define GSTR   260   // gate buffer row stride in floats (256 + 4 bank skew)
#define NTHREADS 512 // 16 waves -> 16 N-tiles of the 256-wide gate matrix

struct __align__(16) Smem {
    _Float16 Ast[3][BTILE][ASTR]; // per layer: cols 0..63 = h (padded), 64..127 = x (padded)
    float    Cst[3][BTILE][64];   // per layer cell state
    float    Gts[BTILE][GSTR];    // gate pre-activations, shared across the 3 layers
};

__device__ __forceinline__ float fast_rcp(float v) {
#if __has_builtin(__builtin_amdgcn_rcpf)
    return __builtin_amdgcn_rcpf(v);   // v_rcp_f32 (TRANS), no IEEE div expansion
#else
    return 1.0f / v;
#endif
}

__device__ __forceinline__ float fast_sigmoid(float v) {
    return fast_rcp(1.0f + __expf(-v));            // v_exp_f32 + v_rcp_f32
}

__device__ __forceinline__ float fast_tanh(float v) {
#if __has_builtin(__builtin_amdgcn_tanhf)
    return __builtin_amdgcn_tanhf(v);              // gfx1250 v_tanh_f32
#else
    float e = __expf(2.0f * v);
    return 1.0f - 2.0f * fast_rcp(e + 1.0f);
#endif
}

// Load a 16x32 f16 A-fragment from an LDS row per the CDNA5 layout:
// lane (idx,half): elements 0..7  -> K = 32*ks + 8*half + 0..7
//                  elements 8..15 -> K = 32*ks + 8*half + 16..23
__device__ __forceinline__ v16h load_a_frag(const _Float16* rowp, int kb) {
    v8h lo = *(const v8h*)(rowp + kb);        // ds_load_b128
    v8h hi = *(const v8h*)(rowp + kb + 16);   // ds_load_b128
    v16h a;
#pragma unroll
    for (int e = 0; e < 8; ++e) { a[e] = lo[e]; a[e + 8] = hi[e]; }
    return a;
}

template<int L, int KS>
__device__ __forceinline__ void layer_step(Smem& sm, const v16h (&Bf)[4], float bias,
                                           int wave, int half, int idx, int tid,
                                           float& xnext, const float* xrow, int t) {
    // ---- GEMM: gates = [h|x] * W^T + b ----
    v8f acc;
#pragma unroll
    for (int r = 0; r < 8; ++r) acc[r] = bias;   // bias depends only on column N

    const _Float16* arow = &sm.Ast[L][idx][0];
    v16h a[KS];
#pragma unroll
    for (int ks = 0; ks < KS; ++ks)              // issue all LDS loads first
        a[ks] = load_a_frag(arow, 32 * ks + 8 * half);
#pragma unroll
    for (int ks = 0; ks < KS; ++ks)
        acc = __builtin_amdgcn_wmma_f32_16x16x32_f16(
                  false, a[ks], false, Bf[ks], (short)0, acc, false, false);

    // D layout: lane l -> N = 16*wave + (l&15); VGPR r -> M = r + 8*(l>>4)
#pragma unroll
    for (int r = 0; r < 8; ++r)
        sm.Gts[r + 8 * half][wave * 16 + idx] = acc[r];
    __syncthreads();

    // ---- elementwise LSTM cell update (i,f,g,o gate order), fully unrolled ----
#pragma unroll
    for (int it = 0; it < (BTILE * 64) / NTHREADS; ++it) {
        const int item = tid + it * NTHREADS;
        const int row = item >> 6, j = item & 63;
        float i = fast_sigmoid(sm.Gts[row][j]);
        float f = fast_sigmoid(sm.Gts[row][64 + j]);
        float g = fast_tanh   (sm.Gts[row][128 + j]);
        float o = fast_sigmoid(sm.Gts[row][192 + j]);
        float c = f * sm.Cst[L][row][j] + i * g;
        sm.Cst[L][row][j] = c;
        float h = o * fast_tanh(c);
        _Float16 hh = (j < HID) ? (_Float16)h : (_Float16)0.0f;
        sm.Ast[L][row][j] = hh;                       // own h for step t+1
        if constexpr (L < 2)
            sm.Ast[L + 1][row][64 + j] = hh;          // next layer's x at this step
    }
    if constexpr (L == 0) {
        // register-pipelined layer-1 input: write x[t+1], prefetch x[t+2]
        if (tid < BTILE) {
            sm.Ast[0][tid][64] = (_Float16)xnext;
            const int t2 = (t + 2 < T_LEN) ? (t + 2) : (T_LEN - 1);
            xnext = xrow[t2];
        }
    }
    __syncthreads();
}

extern "C" __global__ void __launch_bounds__(NTHREADS)
lstm3_fused(const float* __restrict__ x,
            const float* __restrict__ wih1, const float* __restrict__ whh1,
            const float* __restrict__ bih1, const float* __restrict__ bhh1,
            const float* __restrict__ wih2, const float* __restrict__ whh2,
            const float* __restrict__ bih2, const float* __restrict__ bhh2,
            const float* __restrict__ wih3, const float* __restrict__ whh3,
            const float* __restrict__ bih3, const float* __restrict__ bhh3,
            const float* __restrict__ wdense, const float* __restrict__ bdense,
            float* __restrict__ out)
{
    __shared__ Smem sm;
    const int tid  = threadIdx.x;
    const int lane = tid & 31;
    const int wave = tid >> 5;       // 0..15 = N-tile
    const int half = lane >> 4;
    const int idx  = lane & 15;
    const int batch0 = blockIdx.x * BTILE;

    // zero recurrent state + padded lanes
    for (int i = tid; i < 3 * BTILE * ASTR; i += NTHREADS)
        ((_Float16*)sm.Ast)[i] = (_Float16)0.0f;
    for (int i = tid; i < 3 * BTILE * 64; i += NTHREADS)
        ((float*)sm.Cst)[i] = 0.0f;
    __syncthreads();

    // stage x[t=0] and start the x register pipeline (threads 0..15)
    const float* xrow = x + (size_t)(batch0 + (tid & (BTILE - 1))) * T_LEN;
    float xnext = 0.0f;
    if (tid < BTILE) {
        sm.Ast[0][tid][64] = (_Float16)xrow[0];
        xnext = xrow[1];
    }

    // this lane's gate column: n in [0,256), gate g = n>>6, unit j = n&63
    const int  n    = wave * 16 + idx;
    const int  g    = n >> 6;
    const int  j    = n & 63;
    const int  grow = g * HID + j;        // row index in w_ih*/w_hh*/b_* (4H x ...)
    const bool nv   = (j < HID);

    const float bias1 = nv ? bih1[grow] + bhh1[grow] : 0.0f;
    const float bias2 = nv ? bih2[grow] + bhh2[grow] : 0.0f;
    const float bias3 = nv ? bih3[grow] + bhh3[grow] : 0.0f;

    // ---- pack weights into register-resident WMMA B-fragments (one-time) ----
    // B[k][n]: k<64 -> W_hh[grow][k] (k<50), k>=64 -> W_ih[grow][k-64]
    v16h Bf1[4], Bf2[4], Bf3[4];
#pragma unroll
    for (int ks = 0; ks < 4; ++ks) {
#pragma unroll
        for (int e = 0; e < 16; ++e) {
            const int k = 32 * ks + 8 * half + (e & 7) + ((e >> 3) << 4);
            float v1 = 0.0f, v2 = 0.0f, v3 = 0.0f;
            if (nv) {
                if (k < 64) {
                    if (k < HID) {
                        v1 = whh1[grow * HID + k];
                        v2 = whh2[grow * HID + k];
                        v3 = whh3[grow * HID + k];
                    }
                } else {
                    const int kk = k - 64;
                    if (kk == 0) v1 = wih1[grow];          // layer-1 input dim = 1
                    if (kk < HID) {
                        v2 = wih2[grow * HID + kk];
                        v3 = wih3[grow * HID + kk];
                    }
                }
            }
            Bf1[ks][e] = (_Float16)v1;
            Bf2[ks][e] = (_Float16)v2;
            Bf3[ks][e] = (_Float16)v3;
        }
    }
    __syncthreads();

    // ---- sequential scan: all 3 layers fused per timestep ----
    for (int t = 0; t < T_LEN; ++t) {
        layer_step<0, 3>(sm, Bf1, bias1, wave, half, idx, tid, xnext, xrow, t);
        layer_step<1, 4>(sm, Bf2, bias2, wave, half, idx, tid, xnext, xrow, t);
        layer_step<2, 4>(sm, Bf3, bias3, wave, half, idx, tid, xnext, xrow, t);
    }

    // ---- dense head on last hidden state of layer 3 ----
    if (tid < BTILE) {
        float s = 0.0f;
#pragma unroll 1
        for (int jj = 0; jj < HID; ++jj)
            s += (float)sm.Ast[2][tid][jj] * wdense[jj];
        out[batch0 + tid] = s + bdense[0];
    }
}

extern "C" void kernel_launch(void* const* d_in, const int* in_sizes, int n_in,
                              void* d_out, int out_size, void* d_ws, size_t ws_size,
                              hipStream_t stream) {
    (void)in_sizes; (void)n_in; (void)d_ws; (void)ws_size; (void)out_size;
    const float* x     = (const float*)d_in[0];
    const float* wih1  = (const float*)d_in[1];
    const float* whh1  = (const float*)d_in[2];
    const float* bih1  = (const float*)d_in[3];
    const float* bhh1  = (const float*)d_in[4];
    const float* wih2  = (const float*)d_in[5];
    const float* whh2  = (const float*)d_in[6];
    const float* bih2  = (const float*)d_in[7];
    const float* bhh2  = (const float*)d_in[8];
    const float* wih3  = (const float*)d_in[9];
    const float* whh3  = (const float*)d_in[10];
    const float* bih3  = (const float*)d_in[11];
    const float* bhh3  = (const float*)d_in[12];
    const float* wdense = (const float*)d_in[13];
    const float* bdense = (const float*)d_in[14];

    const int B = 512;
    dim3 grid(B / BTILE);   // 32 independent persistent workgroups
    dim3 block(NTHREADS);   // 16 wave32 waves = 16 WMMA N-tiles
    lstm3_fused<<<grid, block, 0, stream>>>(x,
        wih1, whh1, bih1, bhh1,
        wih2, whh2, bih2, bhh2,
        wih3, whh3, bih3, bhh3,
        wdense, bdense, (float*)d_out);
}
